// DSTGCM_3083786518698
// MI455X (gfx1250) — matrix operand order repaired and checked by reading, not compiled, not run on hardware
//
#include <hip/hip_runtime.h>

// ---------------------------------------------------------------------------
// DSTGCM fused pipeline for gfx1250 (MI455X), wave32 + v_wmma_f32_16x16x32_bf16
// ---------------------------------------------------------------------------

typedef __attribute__((ext_vector_type(16))) __bf16 v16bf;
typedef __attribute__((ext_vector_type(8)))  float  v8f;

#define WMMA_BF16(a, b, c) \
  __builtin_amdgcn_wmma_f32_16x16x32_bf16(false, (a), false, (b), (short)0, (c), false, false)

// A-matrix (16x32 bf16) per-lane K map: lanes 0-15 rows M=0..15 hold K 0-15;
// lanes 16-31 same rows hold K 16-31.
__device__ __forceinline__ int mapA(int j, int half) {
  return j + half * 16;
}

// ---------------------------------------------------------------------------
// K1: emb = LayerNorm(node_emb + time_emb) -> bf16   (one wave per 64-wide row)
// ---------------------------------------------------------------------------
__global__ void k_prep_emb(const float* __restrict__ ne, const float* __restrict__ te,
                           const float* __restrict__ gam, const float* __restrict__ bet,
                           __bf16* __restrict__ emb_bf) {
  int lane = threadIdx.x & 31, wave = threadIdx.x >> 5;
  int row = blockIdx.x * 8 + wave;          // 0..32767  (bt*2048 + n)
  int bt = row >> 11, n = row & 2047;
  float a0 = ne[n * 64 + lane]      + te[bt * 64 + lane];
  float a1 = ne[n * 64 + lane + 32] + te[bt * 64 + lane + 32];
  float s = a0 + a1;
  #pragma unroll
  for (int m = 1; m < 32; m <<= 1) s += __shfl_xor(s, m, 32);
  float mu = s * (1.0f / 64.0f);
  float d0 = a0 - mu, d1 = a1 - mu;
  float q = d0 * d0 + d1 * d1;
  #pragma unroll
  for (int m = 1; m < 32; m <<= 1) q += __shfl_xor(q, m, 32);
  float rs = rsqrtf(q * (1.0f / 64.0f) + 1e-12f);
  emb_bf[row * 64 + lane]      = (__bf16)(d0 * rs * gam[lane]      + bet[lane]);
  emb_bf[row * 64 + lane + 32] = (__bf16)(d1 * rs * gam[lane + 32] + bet[lane + 32]);
}

// ---------------------------------------------------------------------------
// K2: x fp32 -> bf16 copy (grid-stride)
// ---------------------------------------------------------------------------
__global__ void k_cvt(const float* __restrict__ x, __bf16* __restrict__ xb, int nElem) {
  int i = blockIdx.x * blockDim.x + threadIdx.x;
  int stride = gridDim.x * blockDim.x;
  for (; i < nElem; i += stride) xb[i] = (__bf16)x[i];
}

// ---------------------------------------------------------------------------
// K3: permute weights_pool [d][k][i][o] -> Bt[col= o*128 + k*64 + i][d] bf16
//     so the W-GEMM B operand is contiguous per lane.
// ---------------------------------------------------------------------------
__global__ void k_wpT(const float* __restrict__ wp, __bf16* __restrict__ Bt) {
  int idx = blockIdx.x * blockDim.x + threadIdx.x;   // 524288 elements
  if (idx >= 64 * 2 * 64 * 64) return;
  int o = idx & 63;
  int i = (idx >> 6) & 63;
  int k = (idx >> 12) & 1;
  int d = idx >> 13;
  int col = o * 128 + k * 64 + i;
  Bt[col * 64 + d] = (__bf16)wp[idx];
}

// ---------------------------------------------------------------------------
// K4: bias[bt][o] = time_emb[bt,:] @ bias_pool  (tiny GEMV)
// ---------------------------------------------------------------------------
__global__ void k_bias(const float* __restrict__ te, const float* __restrict__ bp,
                       float* __restrict__ biasf) {
  int idx = blockIdx.x * blockDim.x + threadIdx.x;   // 1024
  if (idx >= 1024) return;
  int bt = idx >> 6, o = idx & 63;
  float a = 0.0f;
  for (int d = 0; d < 64; ++d) a += te[bt * 64 + d] * bp[d * 64 + o];
  biasf[idx] = a;
}

// ---------------------------------------------------------------------------
// K5: Wt[n][o*128+ki] = node_emb[n,:] @ Bt[:, col]   (WMMA, 16 nodes/block)
//     Output kept bf16 in ws; stays L2-resident (32 MB vs 192 MB L2).
// ---------------------------------------------------------------------------
__global__ void __launch_bounds__(256) k_wt(const float* __restrict__ ne,
                                            const __bf16* __restrict__ Bt,
                                            __bf16* __restrict__ Wt) {
  int lane = threadIdx.x & 31, wave = threadIdx.x >> 5;
  int half = (lane >> 4) & 1, r = lane & 15;
  int nt = blockIdx.x;                               // node tile 0..127
  v16bf a0, a1;
  #pragma unroll
  for (int j = 0; j < 16; ++j) {
    int kA = mapA(j, half);
    a0[j] = (__bf16)ne[(nt * 16 + r) * 64 + kA];
    a1[j] = (__bf16)ne[(nt * 16 + r) * 64 + 32 + kA];
  }
  for (int cb = wave * 64; cb < wave * 64 + 64; ++cb) {   // 512 col chunks / 8 waves
    v8f acc = {};
    v16bf b0, b1;
    const __bf16* bp = Bt + (size_t)(cb * 16 + r) * 64 + half * 16;
    #pragma unroll
    for (int j = 0; j < 16; ++j) { b0[j] = bp[j]; b1[j] = bp[32 + j]; }
    acc = WMMA_BF16(a0, b0, acc);
    acc = WMMA_BF16(a1, b1, acc);
    __bf16* wo = Wt + (size_t)(nt * 16) * 8192 + cb * 16 + r;
    #pragma unroll
    for (int v = 0; v < 8; ++v) wo[(size_t)(v + 8 * half) * 8192] = (__bf16)acc[v];
  }
}

// ---------------------------------------------------------------------------
// K6: flash attention per (b,t): x_g2 = softmax(emb embT) @ x
//     4 waves/block, 16 query rows per wave, 32-key blocks, online softmax.
// ---------------------------------------------------------------------------
__global__ void __launch_bounds__(128) k_attn(const __bf16* __restrict__ emb,
                                              const __bf16* __restrict__ xb,
                                              __bf16* __restrict__ xg2) {
  __shared__ __align__(16) __bf16 e_s[32 * 72];      // keys  (padded: 72*2B=144B rows)
  __shared__ __align__(16) __bf16 x_s[32 * 72];      // values
  __shared__ float p_s[4][16 * 33];                  // per-wave P transpose scratch
  int tid = threadIdx.x, lane = tid & 31, wave = tid >> 5;
  int half = (lane >> 4) & 1, r = lane & 15;
  int bt = blockIdx.x >> 5, qb = blockIdx.x & 31;    // 16 bt x 32 qblocks
  size_t btof = (size_t)bt * 2048 * 64;
  int qrow = qb * 64 + wave * 16 + r;

  v16bf qa0, qa1;                                    // Q 16x64 in A layout
  const __bf16* qp = emb + btof + (size_t)qrow * 64;
  #pragma unroll
  for (int j = 0; j < 16; ++j) { int kA = mapA(j, half); qa0[j] = qp[kA]; qa1[j] = qp[32 + kA]; }

  v8f o0 = {}, o1 = {}, o2 = {}, o3 = {};
  float m[8], l[8];
  #pragma unroll
  for (int v = 0; v < 8; ++v) { m[v] = -1e30f; l[v] = 0.0f; }

  for (int kb = 0; kb < 64; ++kb) {
    __syncthreads();                                 // WAR on e_s/x_s
    {                                                // coop load 32x64 bf16 K and V tiles
      int row = tid >> 2, co = (tid & 3) * 16;
      const uint4* ge = (const uint4*)(emb + btof + (size_t)(kb * 32 + row) * 64 + co);
      const uint4* gx = (const uint4*)(xb  + btof + (size_t)(kb * 32 + row) * 64 + co);
      uint4* le = (uint4*)(e_s + row * 72 + co);
      uint4* lx = (uint4*)(x_s + row * 72 + co);
      le[0] = ge[0]; le[1] = ge[1];
      lx[0] = gx[0]; lx[1] = gx[1];
    }
    __syncthreads();

    // S = Q @ K^T : two 16-column chunks, K-dim 64 in two WMMA steps each
    v8f s0 = {}, s1 = {};
    {
      v16bf b0, b1;
      const __bf16* bp = e_s + (r) * 72 + half * 16;
      #pragma unroll
      for (int j = 0; j < 16; ++j) { b0[j] = bp[j]; b1[j] = bp[32 + j]; }
      s0 = WMMA_BF16(qa0, b0, s0);
      s0 = WMMA_BF16(qa1, b1, s0);
      bp = e_s + (16 + r) * 72 + half * 16;
      #pragma unroll
      for (int j = 0; j < 16; ++j) { b0[j] = bp[j]; b1[j] = bp[32 + j]; }
      s1 = WMMA_BF16(qa0, b0, s1);
      s1 = WMMA_BF16(qa1, b1, s1);
    }

    // online softmax over the 32 keys; row M=v+8*half lives on one 16-lane half
    float scale[8];
    #pragma unroll
    for (int v = 0; v < 8; ++v) {
      float mx = fmaxf(s0[v], s1[v]);
      mx = fmaxf(mx, __shfl_xor(mx, 1, 32)); mx = fmaxf(mx, __shfl_xor(mx, 2, 32));
      mx = fmaxf(mx, __shfl_xor(mx, 4, 32)); mx = fmaxf(mx, __shfl_xor(mx, 8, 32));
      float mn = fmaxf(m[v], mx);
      float sc = __expf(m[v] - mn);
      float p0 = __expf(s0[v] - mn);
      float p1 = __expf(s1[v] - mn);
      float rs = p0 + p1;
      rs += __shfl_xor(rs, 1, 32); rs += __shfl_xor(rs, 2, 32);
      rs += __shfl_xor(rs, 4, 32); rs += __shfl_xor(rs, 8, 32);
      l[v] = l[v] * sc + rs;
      m[v] = mn;
      scale[v] = sc;
      s0[v] = p0; s1[v] = p1;
      o0[v] *= sc; o1[v] *= sc; o2[v] *= sc; o3[v] *= sc;
    }

    // P: C-layout -> LDS -> A-layout bf16 (wave-private region, in-order DS)
    float* ps = p_s[wave];
    #pragma unroll
    for (int v = 0; v < 8; ++v) {
      ps[(v + 8 * half) * 33 + r]      = s0[v];
      ps[(v + 8 * half) * 33 + 16 + r] = s1[v];
    }
    v16bf pa;
    #pragma unroll
    for (int j = 0; j < 16; ++j) pa[j] = (__bf16)ps[r * 33 + mapA(j, half)];

    // O += P @ V : four 16-col output chunks
    {
      v16bf vb;
      #pragma unroll
      for (int j = 0; j < 16; ++j) vb[j] = x_s[(half * 16 + j) * 72 + r];
      o0 = WMMA_BF16(pa, vb, o0);
      #pragma unroll
      for (int j = 0; j < 16; ++j) vb[j] = x_s[(half * 16 + j) * 72 + 16 + r];
      o1 = WMMA_BF16(pa, vb, o1);
      #pragma unroll
      for (int j = 0; j < 16; ++j) vb[j] = x_s[(half * 16 + j) * 72 + 32 + r];
      o2 = WMMA_BF16(pa, vb, o2);
      #pragma unroll
      for (int j = 0; j < 16; ++j) vb[j] = x_s[(half * 16 + j) * 72 + 48 + r];
      o3 = WMMA_BF16(pa, vb, o3);
    }
  }

  __bf16* op = xg2 + btof + (size_t)(qb * 64 + wave * 16) * 64;
  #pragma unroll
  for (int v = 0; v < 8; ++v) {
    float inv = 1.0f / l[v];
    int rr = v + 8 * half;
    op[rr * 64 + r]      = (__bf16)(o0[v] * inv);
    op[rr * 64 + 16 + r] = (__bf16)(o1[v] * inv);
    op[rr * 64 + 32 + r] = (__bf16)(o2[v] * inv);
    op[rr * 64 + 48 + r] = (__bf16)(o3[v] * inv);
  }
}

// ---------------------------------------------------------------------------
// K7: out[bt][n][o] = [x | x_g2][bt][n][:] @ W[n] + bias[bt][o]
//     One wave per node: 16(bt) x 128(ki) x 64(o) = 16 WMMAs.
// ---------------------------------------------------------------------------
__global__ void __launch_bounds__(256) k_gconv(const __bf16* __restrict__ xb,
                                               const __bf16* __restrict__ xg2,
                                               const __bf16* __restrict__ Wt,
                                               const float* __restrict__ biasf,
                                               float* __restrict__ out) {
  int lane = threadIdx.x & 31, wave = threadIdx.x >> 5;
  int half = (lane >> 4) & 1, r = lane & 15;     // A row M = bt = r
  int n = blockIdx.x * 8 + wave;
  v16bf a0, a1, a2, a3;
  const __bf16* xp  = xb  + ((size_t)r * 2048 + n) * 64;
  const __bf16* x2p = xg2 + ((size_t)r * 2048 + n) * 64;
  #pragma unroll
  for (int j = 0; j < 16; ++j) {
    int kA = mapA(j, half);
    a0[j] = xp[kA];  a1[j] = xp[32 + kA];
    a2[j] = x2p[kA]; a3[j] = x2p[32 + kA];
  }
  const __bf16* wp = Wt + (size_t)n * 8192;
  #pragma unroll
  for (int cc = 0; cc < 4; ++cc) {
    v8f acc = {};
    const __bf16* bp = wp + (size_t)(cc * 16 + r) * 128 + half * 16;
    v16bf b;
    #pragma unroll
    for (int j = 0; j < 16; ++j) b[j] = bp[j];
    acc = WMMA_BF16(a0, b, acc);
    #pragma unroll
    for (int j = 0; j < 16; ++j) b[j] = bp[32 + j];
    acc = WMMA_BF16(a1, b, acc);
    #pragma unroll
    for (int j = 0; j < 16; ++j) b[j] = bp[64 + j];
    acc = WMMA_BF16(a2, b, acc);
    #pragma unroll
    for (int j = 0; j < 16; ++j) b[j] = bp[96 + j];
    acc = WMMA_BF16(a3, b, acc);
    #pragma unroll
    for (int v = 0; v < 8; ++v) {
      int btr = v + 8 * half;
      out[((size_t)btr * 2048 + n) * 64 + cc * 16 + r] = acc[v] + biasf[btr * 64 + cc * 16 + r];
    }
  }
}

// ---------------------------------------------------------------------------
// Host launcher
// ---------------------------------------------------------------------------
extern "C" void kernel_launch(void* const* d_in, const int* in_sizes, int n_in,
                              void* d_out, int out_size, void* d_ws, size_t ws_size,
                              hipStream_t stream) {
  const float* x     = (const float*)d_in[0];   // (8,2,2048,64)
  const float* ne    = (const float*)d_in[1];   // (2048,64)
  const float* te    = (const float*)d_in[2];   // (8,2,64)
  const float* wpool = (const float*)d_in[3];   // (64,2,64,64)
  const float* bpool = (const float*)d_in[4];   // (64,64)
  const float* gam   = (const float*)d_in[5];   // (64,)
  const float* bet   = (const float*)d_in[6];   // (64,)
  float* out = (float*)d_out;

  char* ws = (char*)d_ws;
  const size_t MB = 1024 * 1024;
  __bf16* emb_bf = (__bf16*)(ws + 0);          //  4 MB: (16,2048,64) bf16
  __bf16* x_bf   = (__bf16*)(ws + 4 * MB);     //  4 MB: (16,2048,64) bf16
  __bf16* Bt_bf  = (__bf16*)(ws + 8 * MB);     //  1 MB: (8192,64)    bf16 permuted pool
  __bf16* Wt_bf  = (__bf16*)(ws + 12 * MB);    // 32 MB: (2048,64,128) bf16 per-node W
  __bf16* xg2_bf = (__bf16*)(ws + 44 * MB);    //  4 MB: (16,2048,64) bf16
  float*  bias_f = (float*)(ws + 48 * MB);     //  4 KB: (16,64)

  k_prep_emb<<<4096, 256, 0, stream>>>(ne, te, gam, bet, emb_bf);
  k_cvt<<<2048, 256, 0, stream>>>(x, x_bf, 16 * 2048 * 64);
  k_wpT<<<2048, 256, 0, stream>>>(wpool, Bt_bf);
  k_bias<<<4, 256, 0, stream>>>(te, bpool, bias_f);
  k_wt<<<128, 256, 0, stream>>>(ne, Bt_bf, Wt_bf);
  k_attn<<<512, 128, 0, stream>>>(emb_bf, x_bf, xg2_bf);
  k_gconv<<<256, 256, 0, stream>>>(x_bf, xg2_bf, Wt_bf, bias_f, out);
}
